// MultiheadAttention_49495203119191
// MI455X (gfx1250) — matrix operand order repaired and checked by reading, not compiled, not run on hardware
//
#include <hip/hip_runtime.h>
#include <hip/hip_bf16.h>

// ---------------------------------------------------------------------------
// MI455X (gfx1250) fused multi-head attention, bf16 WMMA path.
// B=2, S=2048, E=1024, H=16, D=64. ~69 GFLOP vs ~100MB traffic -> compute
// bound; v_wmma_f32_16x16x32_bf16 (K=32/instr, f32 accum) is the dense path.
// Round-4: fully unroll the ping-pong K loop. Round 3 showed the rolled loop
// created accumulator phis -> v_mov_b64 copies + WMMA->VALU hazard v_nops in
// the inner loop. Unrolling removes the phis while keeping the pipelined
// (partial s_wait_loadcnt) load schedule.
// ---------------------------------------------------------------------------

#define EMB 1024
#define SEQ 2048
#define NH 16
#define HD 64

typedef __attribute__((ext_vector_type(16))) __bf16 v16bf;
typedef __attribute__((ext_vector_type(8)))  __bf16 v8bf;
typedef __attribute__((ext_vector_type(8)))  float  v8f;

// ---- WMMA operand loaders (ISA 7.12.2 layouts, wave32) --------------------

// A matrix 16x32 (MxK), bf16, from row-major src (leading dim ld, elements).
__device__ __forceinline__ v16bf load_a(const __bf16* src, int ld) {
  const int lane = threadIdx.x & 31;
  const __bf16* p = src + (size_t)(lane & 15) * ld + ((lane >> 4) << 3);
  v8bf lo = *(const v8bf*)p;         // K = kb .. kb+7
  v8bf hi = *(const v8bf*)(p + 16);  // K = kb+16 .. kb+23
  v16bf a;
#pragma unroll
  for (int i = 0; i < 8; ++i) { a[i] = lo[i]; a[i + 8] = hi[i]; }
  return a;
}

// B matrix 32x16 (KxN) with B[k][n] = src[n][k] (src row-major [N x K]).
__device__ __forceinline__ v16bf load_bt(const __bf16* src, int ld) {
  const int lane = threadIdx.x & 31;
  const __bf16* p = src + (size_t)(lane & 15) * ld + ((lane >> 4) << 3);
  v8bf lo = *(const v8bf*)p;
  v8bf hi = *(const v8bf*)(p + 16);
  v16bf b;
#pragma unroll
  for (int i = 0; i < 8; ++i) { b[i] = lo[i]; b[i + 8] = hi[i]; }
  return b;
}

__device__ __forceinline__ v8f wmma_bf16(v16bf a, v16bf b, v8f c) {
  return __builtin_amdgcn_wmma_f32_16x16x32_bf16(false, a, false, b, (short)0, c,
                                                 false, false);
}

// 16-lane (half-wave) reductions: C-layout row lives in one VGPR across 16 lanes
__device__ __forceinline__ float hmax16(float v) {
#pragma unroll
  for (int m = 1; m < 16; m <<= 1) v = fmaxf(v, __shfl_xor(v, m, 32));
  return v;
}
__device__ __forceinline__ float hsum16(float v) {
#pragma unroll
  for (int m = 1; m < 16; m <<= 1) v += __shfl_xor(v, m, 32);
  return v;
}

// ---- GEMM fragment group: 32x64 tile, one k-step of 32 --------------------
struct Frag {
  v16bf a0, a1;          // rows m0..m0+15, m0+16..m0+31
  v16bf b0, b1, b2, b3;  // cols n0+0..15, 16..31, 32..47, 48..63
};

__device__ __forceinline__ void load_group(Frag& f, const __bf16* arow,
                                           const __bf16* wbase, int k0) {
  f.a0 = load_a(arow + k0, EMB);
  f.a1 = load_a(arow + (size_t)16 * EMB + k0, EMB);
  f.b0 = load_bt(wbase + k0, EMB);
  f.b1 = load_bt(wbase + (size_t)16 * EMB + k0, EMB);
  f.b2 = load_bt(wbase + (size_t)32 * EMB + k0, EMB);
  f.b3 = load_bt(wbase + (size_t)48 * EMB + k0, EMB);
}

__device__ __forceinline__ void mma_group(v8f acc[8], const Frag& f) {
  acc[0] = wmma_bf16(f.a0, f.b0, acc[0]);
  acc[4] = wmma_bf16(f.a1, f.b0, acc[4]);
  acc[1] = wmma_bf16(f.a0, f.b1, acc[1]);
  acc[5] = wmma_bf16(f.a1, f.b1, acc[5]);
  acc[2] = wmma_bf16(f.a0, f.b2, acc[2]);
  acc[6] = wmma_bf16(f.a1, f.b2, acc[6]);
  acc[3] = wmma_bf16(f.a0, f.b3, acc[3]);
  acc[7] = wmma_bf16(f.a1, f.b3, acc[7]);
}

// ping-pong mainloop over K = EMB in steps of 32, two buffers, fully unrolled
__device__ __forceinline__ void gemm_32x64(v8f acc[8], const __bf16* arow,
                                           const __bf16* wbase) {
  Frag f0, f1;
  load_group(f0, arow, wbase, 0);
#pragma unroll
  for (int k0 = 0; k0 < EMB - 64; k0 += 64) {
    load_group(f1, arow, wbase, k0 + 32);
    mma_group(acc, f0);
    load_group(f0, arow, wbase, k0 + 64);
    mma_group(acc, f1);
  }
  load_group(f1, arow, wbase, EMB - 32);
  mma_group(acc, f0);
  mma_group(acc, f1);
}

// ---- kernel 0: f32 -> bf16 convert ----------------------------------------
__global__ void cvt_f32_bf16(const float* __restrict__ src,
                             __bf16* __restrict__ dst, int n) {
  int i = (blockIdx.x * blockDim.x + threadIdx.x) * 4;
  if (i < n) {
    float4 v = *(const float4*)(src + i);
    dst[i + 0] = (__bf16)v.x;
    dst[i + 1] = (__bf16)v.y;
    dst[i + 2] = (__bf16)v.z;
    dst[i + 3] = (__bf16)v.w;
  }
}

// ---- kernel 1: QKV projection + per-head LayerNorm ------------------------
// grid: (BS/256, NH, 3)  block: 256 (8 waves, each wave -> 32 rows x 64 cols)
__global__ __launch_bounds__(256) void qkv_ln(
    const __bf16* __restrict__ xb, const __bf16* __restrict__ wq,
    const __bf16* __restrict__ wk, const __bf16* __restrict__ wv,
    const float* __restrict__ bq, const float* __restrict__ bk,
    const float* __restrict__ bv, const float* __restrict__ qnw,
    const float* __restrict__ qnb, const float* __restrict__ knw,
    const float* __restrict__ knb, __bf16* __restrict__ qo,
    __bf16* __restrict__ ko, __bf16* __restrict__ vT) {
  const int lane = threadIdx.x & 31;
  const int w = threadIdx.x >> 5;
  const int nl = lane & 15;
  const int half8 = (lane >> 4) * 8;
  const int m0 = (blockIdx.x * 8 + w) * 32;  // row tile in [B*S]
  const int h = blockIdx.y;
  const int z = blockIdx.z;  // 0=Q 1=K 2=V
  const int n0 = h * HD;

  const __bf16* W = (z == 0) ? wq : (z == 1) ? wk : wv;
  const __bf16* arow = xb + (size_t)m0 * EMB;
  const __bf16* wbase = W + (size_t)n0 * EMB;

  v8f acc[8] = {};
  gemm_32x64(acc, arow, wbase);

  const float* bias = (z == 0) ? bq : (z == 1) ? bk : bv;
#pragma unroll
  for (int blk = 0; blk < 4; ++blk) {
    float bb = bias[n0 + blk * 16 + nl];
#pragma unroll
    for (int r = 0; r < 8; ++r) {
      acc[blk][r] += bb;
      acc[4 + blk][r] += bb;
    }
  }

  if (z < 2) {
    // LayerNorm over the 64 head columns of each row (eps 1e-5)
#pragma unroll
    for (int rg = 0; rg < 2; ++rg)
#pragma unroll
      for (int r = 0; r < 8; ++r) {
        float s = acc[rg * 4 + 0][r] + acc[rg * 4 + 1][r] +
                  acc[rg * 4 + 2][r] + acc[rg * 4 + 3][r];
        float mu = hsum16(s) * (1.f / 64.f);
        float vv = 0.f;
#pragma unroll
        for (int blk = 0; blk < 4; ++blk) {
          float d = acc[rg * 4 + blk][r] - mu;
          vv += d * d;
        }
        vv = hsum16(vv) * (1.f / 64.f);
        float inv = rsqrtf(vv + 1e-5f);
#pragma unroll
        for (int blk = 0; blk < 4; ++blk)
          acc[rg * 4 + blk][r] = (acc[rg * 4 + blk][r] - mu) * inv;
      }
    const float* lw = (z == 0) ? qnw : knw;
    const float* lb = (z == 0) ? qnb : knb;
    __bf16* dst = (z == 0) ? qo : ko;
#pragma unroll
    for (int blk = 0; blk < 4; ++blk) {
      float sw = lw[blk * 16 + nl], sb = lb[blk * 16 + nl];
#pragma unroll
      for (int rg = 0; rg < 2; ++rg)
#pragma unroll
        for (int r = 0; r < 8; ++r) {
          int m = m0 + rg * 16 + r + half8;
          int bb = m >> 11, srow = m & (SEQ - 1);
          dst[(((size_t)(bb * NH + h)) * SEQ + srow) * HD + blk * 16 + nl] =
              (__bf16)(acc[rg * 4 + blk][r] * sw + sb);
        }
    }
  } else {
    // V stored transposed [B,H,D,S] so PV WMMA B-operand loads are contiguous
#pragma unroll
    for (int blk = 0; blk < 4; ++blk)
#pragma unroll
      for (int rg = 0; rg < 2; ++rg)
#pragma unroll
        for (int r = 0; r < 8; ++r) {
          int m = m0 + rg * 16 + r + half8;
          int bb = m >> 11, srow = m & (SEQ - 1);
          int d = blk * 16 + nl;
          vT[(((size_t)(bb * NH + h)) * HD + d) * SEQ + srow] =
              (__bf16)acc[rg * 4 + blk][r];
        }
  }
}

// ---- kernel 2: flash attention, one 16-row query tile per wave ------------
// grid: (SEQ/16/8, B*NH)  block: 256
__global__ __launch_bounds__(256) void attn_fa(const __bf16* __restrict__ q,
                                               const __bf16* __restrict__ kk,
                                               const __bf16* __restrict__ vT,
                                               __bf16* __restrict__ ao) {
  __shared__ __align__(16) __bf16 pbuf[8][16 * 32];  // per-wave P staging (1KB)
  const int lane = threadIdx.x & 31;
  const int w = threadIdx.x >> 5;
  const int nl = lane & 15;
  const int half8 = (lane >> 4) * 8;
  const int bh = blockIdx.y;
  const int b = bh >> 4, h = bh & 15;
  const int i0 = (blockIdx.x * 8 + w) * 16;

  const __bf16* qh = q + (size_t)bh * SEQ * HD;
  const __bf16* kh = kk + (size_t)bh * SEQ * HD;
  const __bf16* vh = vT + (size_t)bh * HD * SEQ;

  v16bf qa0 = load_a(qh + (size_t)i0 * HD, HD);
  v16bf qa1 = load_a(qh + (size_t)i0 * HD + 32, HD);
#pragma unroll
  for (int t = 0; t < 16; ++t) {  // fold D^-0.5 = 0.125 into Q fragment
    qa0[t] = (__bf16)((float)qa0[t] * 0.125f);
    qa1[t] = (__bf16)((float)qa1[t] * 0.125f);
  }

  float m_run[8], l_run[8];
#pragma unroll
  for (int r = 0; r < 8; ++r) { m_run[r] = -1e30f; l_run[r] = 0.f; }
  v8f accO[4] = {};
  __bf16* pw = &pbuf[w][0];

  // pipeline prologue: K fragments for j0 = 0
  v16bf kb[4];
  kb[0] = load_bt(kh, HD);
  kb[1] = load_bt(kh + 32, HD);
  kb[2] = load_bt(kh + 16 * HD, HD);
  kb[3] = load_bt(kh + 16 * HD + 32, HD);

  for (int j0 = 0; j0 < SEQ; j0 += 32) {
    v8f s0 = {}, s1 = {};
    s0 = wmma_bf16(qa0, kb[0], s0);
    s0 = wmma_bf16(qa1, kb[1], s0);
    s1 = wmma_bf16(qa0, kb[2], s1);
    s1 = wmma_bf16(qa1, kb[3], s1);

    // issue V loads for this step + K loads for the next step now, so their
    // latency hides behind the softmax VALU/shuffle section below
    v16bf vb[4];
#pragma unroll
    for (int blk = 0; blk < 4; ++blk)
      vb[blk] = load_bt(vh + (size_t)(blk * 16) * SEQ + j0, SEQ);
    const int jn = (j0 + 32 < SEQ) ? j0 + 32 : j0;
    kb[0] = load_bt(kh + (size_t)jn * HD, HD);
    kb[1] = load_bt(kh + (size_t)jn * HD + 32, HD);
    kb[2] = load_bt(kh + (size_t)(jn + 16) * HD, HD);
    kb[3] = load_bt(kh + (size_t)(jn + 16) * HD + 32, HD);

#pragma unroll
    for (int r = 0; r < 8; ++r) {
      float mx = hmax16(fmaxf(s0[r], s1[r]));
      float mnew = fmaxf(m_run[r], mx);
      float alpha = __expf(m_run[r] - mnew);
      float p0 = __expf(s0[r] - mnew);
      float p1 = __expf(s1[r] - mnew);
      float rs = hsum16(p0 + p1);
      l_run[r] = l_run[r] * alpha + rs;
      m_run[r] = mnew;
      accO[0][r] *= alpha;
      accO[1][r] *= alpha;
      accO[2][r] *= alpha;
      accO[3][r] *= alpha;
      // C-layout -> LDS row-major P tile [16 x 32]
      pw[(r + half8) * 32 + nl] = (__bf16)p0;
      pw[(r + half8) * 32 + 16 + nl] = (__bf16)p1;
    }
    asm volatile("s_wait_dscnt 0" ::: "memory");
    v16bf pa = load_a(pw, 32);  // ds_load_b128 x2: A-layout P fragment
#pragma unroll
    for (int blk = 0; blk < 4; ++blk)
      accO[blk] = wmma_bf16(pa, vb[blk], accO[blk]);
  }

#pragma unroll
  for (int blk = 0; blk < 4; ++blk)
#pragma unroll
    for (int r = 0; r < 8; ++r) {
      float o = accO[blk][r] / l_run[r];
      int srow = i0 + r + half8;
      ao[((size_t)(b * SEQ + srow)) * EMB + h * HD + blk * 16 + nl] = (__bf16)o;
    }
}

// ---- kernel 3: output projection ------------------------------------------
// grid: (BS/256, EMB/64)  block: 256 (8 waves, each wave -> 32 rows x 64 cols)
__global__ __launch_bounds__(256) void oproj(const __bf16* __restrict__ ao,
                                             const __bf16* __restrict__ wo,
                                             const float* __restrict__ bo,
                                             float* __restrict__ out) {
  const int lane = threadIdx.x & 31;
  const int w = threadIdx.x >> 5;
  const int nl = lane & 15;
  const int half8 = (lane >> 4) * 8;
  const int m0 = (blockIdx.x * 8 + w) * 32;
  const int n0 = blockIdx.y * 64;
  const __bf16* arow = ao + (size_t)m0 * EMB;
  const __bf16* wbase = wo + (size_t)n0 * EMB;

  v8f acc[8] = {};
  gemm_32x64(acc, arow, wbase);

#pragma unroll
  for (int blk = 0; blk < 4; ++blk) {
    float bb = bo[n0 + blk * 16 + nl];
#pragma unroll
    for (int rg = 0; rg < 2; ++rg)
#pragma unroll
      for (int r = 0; r < 8; ++r)
        out[(size_t)(m0 + rg * 16 + r + half8) * EMB + n0 + blk * 16 + nl] =
            acc[rg * 4 + blk][r] + bb;
  }
}

// ---------------------------------------------------------------------------
extern "C" void kernel_launch(void* const* d_in, const int* in_sizes, int n_in,
                              void* d_out, int out_size, void* d_ws,
                              size_t ws_size, hipStream_t stream) {
  (void)n_in; (void)out_size; (void)ws_size;
  const float* x = (const float*)d_in[0];
  const float* Wq = (const float*)d_in[1];
  const float* bq = (const float*)d_in[2];
  const float* Wk = (const float*)d_in[3];
  const float* bk = (const float*)d_in[4];
  const float* Wv = (const float*)d_in[5];
  const float* bv = (const float*)d_in[6];
  const float* Wo = (const float*)d_in[7];
  const float* bo = (const float*)d_in[8];
  const float* qnw = (const float*)d_in[9];
  const float* qnb = (const float*)d_in[10];
  const float* knw = (const float*)d_in[11];
  const float* knb = (const float*)d_in[12];

  const int BS = in_sizes[0] / EMB;  // B*S = 4096
  const int B = BS / SEQ;            // 2

  // workspace layout (bf16): 48 MB total
  __bf16* xb = (__bf16*)d_ws;                  // BS*E
  __bf16* wqb = xb + (size_t)BS * EMB;         // E*E
  __bf16* wkb = wqb + (size_t)EMB * EMB;
  __bf16* wvb = wkb + (size_t)EMB * EMB;
  __bf16* wob = wvb + (size_t)EMB * EMB;
  __bf16* qb = wob + (size_t)EMB * EMB;        // [B,H,S,D]
  __bf16* kb = qb + (size_t)BS * EMB;          // [B,H,S,D]
  __bf16* vT = kb + (size_t)BS * EMB;          // [B,H,D,S]
  __bf16* aob = vT + (size_t)BS * EMB;         // [B,S,H*D]

  {
    int n = BS * EMB;
    cvt_f32_bf16<<<(n / 4 + 255) / 256, 256, 0, stream>>>(x, xb, n);
    n = EMB * EMB;
    cvt_f32_bf16<<<(n / 4 + 255) / 256, 256, 0, stream>>>(Wq, wqb, n);
    cvt_f32_bf16<<<(n / 4 + 255) / 256, 256, 0, stream>>>(Wk, wkb, n);
    cvt_f32_bf16<<<(n / 4 + 255) / 256, 256, 0, stream>>>(Wv, wvb, n);
    cvt_f32_bf16<<<(n / 4 + 255) / 256, 256, 0, stream>>>(Wo, wob, n);
  }

  qkv_ln<<<dim3(BS / 256, NH, 3), 256, 0, stream>>>(
      xb, wqb, wkb, wvb, bq, bk, bv, qnw, qnb, knw, knb, qb, kb, vT);

  attn_fa<<<dim3(SEQ / 16 / 8, B * NH), 256, 0, stream>>>(qb, kb, vT, aob);

  oproj<<<dim3(BS / 256, EMB / 64), 256, 0, stream>>>(aob, wob, bo,
                                                      (float*)d_out);
}
